// MultiHeadAttentionFast_10333691314637
// MI455X (gfx1250) — compile-verified, matrix operand order checked by reading, not run on hardware
//
#include <hip/hip_runtime.h>

// ---------------------------------------------------------------------------
// MI455X (gfx1250) multi-head attention, bf16 WMMA (16x16x32), fp32 accum.
// Wave32. A/B fragment layout per CDNA5 ISA 7.12.2:
//   lane l (r = l&15, half = l>>4) holds row/col r; bf16 elems j=0..7 -> K =
//   half*8 + j (contiguous), j=8..15 -> K = 16 + half*8 + (j-8) (contiguous).
//   => each fragment = two 16-byte loads from a row of the operand.
// C/D v8f layout: VGPR i, lanes 0-15: (M=i, N=lane); lanes 16-31: (M=i+8).
// ---------------------------------------------------------------------------

#define DEV __device__ __forceinline__

typedef __attribute__((ext_vector_type(16))) __bf16 v16bf;
typedef __attribute__((ext_vector_type(8)))  float  v8f;

static constexpr int Sq = 4096;   // sequence
static constexpr int Dm = 1024;   // model dim
static constexpr int Hh = 16;     // heads
static constexpr int Dk = 64;     // head dim

union FragU { v16bf v; uint4 u[2]; };

DEV v16bf load_frag(const __bf16* rowp, int half, int k0) {
  FragU f;
  f.u[0] = *(const uint4*)(rowp + k0 + half * 8);
  f.u[1] = *(const uint4*)(rowp + k0 + 16 + half * 8);
  return f.v;
}

DEV v8f wmma_bf16(v16bf a, v16bf b, v8f c) {
  return __builtin_amdgcn_wmma_f32_16x16x32_bf16(
      /*neg_a=*/false, a, /*neg_b=*/false, b,
      /*c_mod=*/(short)0, c, /*reuse_a=*/false, /*reuse_b=*/false);
}

DEV v8f vzero() {
  v8f z;
#pragma unroll
  for (int i = 0; i < 8; ++i) z[i] = 0.f;
  return z;
}

// ---------------------------------------------------------------------------
// elementwise converts
// ---------------------------------------------------------------------------
__global__ void cvt_f32_bf16(const float* __restrict__ in,
                             __bf16* __restrict__ out, int n) {
  int i = blockIdx.x * blockDim.x + threadIdx.x;
  if (i < n) out[i] = (__bf16)in[i];
}

// out[n*1024 + k] = (bf16) in[k*Nd + n]   (Kdim fixed at 1024)
__global__ void transpose_cvt(const float* __restrict__ in,
                              __bf16* __restrict__ out, int Nd, int total) {
  int i = blockIdx.x * blockDim.x + threadIdx.x;
  if (i < total) {
    int n = i >> 10;
    int k = i & 1023;
    out[i] = (__bf16)in[(size_t)k * Nd + n];
  }
}

// ---------------------------------------------------------------------------
// shared GEMM mainloop: 32x64 output tile per wave, K = 1024
// ---------------------------------------------------------------------------
DEV void gemm_core(const __bf16* x0, const __bf16* x1,
                   const __bf16* w0, const __bf16* w1,
                   const __bf16* w2, const __bf16* w3,
                   int half, v8f (&acc)[2][4]) {
  for (int k0 = 0; k0 < Dm; k0 += 32) {
    // global_prefetch_b8: pull rows two unrolled steps ahead into the caches
    if (k0 + 64 < Dm) {
      __builtin_prefetch(x0 + k0 + 64, 0, 3);
      __builtin_prefetch(x1 + k0 + 64, 0, 3);
      __builtin_prefetch(w0 + k0 + 64, 0, 3);
      __builtin_prefetch(w1 + k0 + 64, 0, 3);
      __builtin_prefetch(w2 + k0 + 64, 0, 3);
      __builtin_prefetch(w3 + k0 + 64, 0, 3);
    }
    v16bf a0 = load_frag(x0, half, k0);
    v16bf a1 = load_frag(x1, half, k0);
    v16bf b0 = load_frag(w0, half, k0);
    v16bf b1 = load_frag(w1, half, k0);
    v16bf b2 = load_frag(w2, half, k0);
    v16bf b3 = load_frag(w3, half, k0);
    acc[0][0] = wmma_bf16(a0, b0, acc[0][0]);
    acc[0][1] = wmma_bf16(a0, b1, acc[0][1]);
    acc[0][2] = wmma_bf16(a0, b2, acc[0][2]);
    acc[0][3] = wmma_bf16(a0, b3, acc[0][3]);
    acc[1][0] = wmma_bf16(a1, b0, acc[1][0]);
    acc[1][1] = wmma_bf16(a1, b1, acc[1][1]);
    acc[1][2] = wmma_bf16(a1, b2, acc[1][2]);
    acc[1][3] = wmma_bf16(a1, b3, acc[1][3]);
  }
}

// ---------------------------------------------------------------------------
// QKV projection: qkv[8192,3072] = Xbf16 @ WqkvT' ; scatter epilogue into
// Q bf16 [B,H,S,64], K bf16 [B,H,S,64], V bf16 transposed [B,H,64,S].
// ---------------------------------------------------------------------------
__global__ __launch_bounds__(256) void gemm_qkv(
    const __bf16* __restrict__ X, const __bf16* __restrict__ Wt,
    const float* __restrict__ bias,
    __bf16* __restrict__ Qw, __bf16* __restrict__ Kw,
    __bf16* __restrict__ Vtw) {
  int lane = threadIdx.x & 31, wave = threadIdx.x >> 5;
  int r = lane & 15, half = lane >> 4;
  int m0 = blockIdx.y * 256 + wave * 32;
  int n0 = blockIdx.x * 64;

  v8f acc[2][4];
#pragma unroll
  for (int a = 0; a < 2; ++a)
#pragma unroll
    for (int t = 0; t < 4; ++t) acc[a][t] = vzero();

  gemm_core(X + (size_t)(m0 + r) * Dm, X + (size_t)(m0 + 16 + r) * Dm,
            Wt + (size_t)(n0 + r) * Dm, Wt + (size_t)(n0 + 16 + r) * Dm,
            Wt + (size_t)(n0 + 32 + r) * Dm, Wt + (size_t)(n0 + 48 + r) * Dm,
            half, acc);

#pragma unroll
  for (int mt = 0; mt < 2; ++mt) {
#pragma unroll
    for (int t = 0; t < 4; ++t) {
      int n = n0 + t * 16 + r;
      float bs = bias[n];
      int sec = n >> 10;          // 0=Q 1=K 2=V
      int h = (n >> 6) & 15;
      int d = n & 63;
#pragma unroll
      for (int i = 0; i < 8; ++i) {
        int mm = m0 + mt * 16 + 8 * half + i;
        int b = mm >> 12;
        int s = mm & (Sq - 1);
        __bf16 bv = (__bf16)(acc[mt][t][i] + bs);
        size_t bh = (size_t)(b * Hh + h);
        if (sec == 0)      Qw[(bh * Sq + s) * Dk + d] = bv;
        else if (sec == 1) Kw[(bh * Sq + s) * Dk + d] = bv;
        else               Vtw[(bh * Dk + d) * Sq + s] = bv;
      }
    }
  }
}

// ---------------------------------------------------------------------------
// Flash attention: 1 wave = 16 q rows; causal loop over 32-key blocks.
// The inner loop is a serial chain (scores -> softmax -> P@V), so prefetch
// next key-block's K/V rows one block ahead to hide L2/HBM latency.
// ---------------------------------------------------------------------------
__global__ __launch_bounds__(256) void attn_fwd(
    const __bf16* __restrict__ Q, const __bf16* __restrict__ K,
    const __bf16* __restrict__ Vt, __bf16* __restrict__ AO) {
  __shared__ __align__(16) __bf16 lds_p[8][16 * 32];

  int lane = threadIdx.x & 31, wave = threadIdx.x >> 5;
  int r = lane & 15, half = lane >> 4;
  int bh = blockIdx.y;
  int b = bh >> 4, h = bh & 15;
  const __bf16* Qp = Q  + (size_t)bh * Sq * Dk;
  const __bf16* Kp = K  + (size_t)bh * Sq * Dk;
  const __bf16* Vp = Vt + (size_t)bh * Dk * Sq;
  int qb = blockIdx.x * 128 + wave * 16;

  v16bf qa0 = load_frag(Qp + (size_t)(qb + r) * Dk, half, 0);
  v16bf qa1 = load_frag(Qp + (size_t)(qb + r) * Dk, half, 32);

  float rowmax[8], rowsum[8];
#pragma unroll
  for (int i = 0; i < 8; ++i) { rowmax[i] = -__builtin_inff(); rowsum[i] = 0.f; }
  v8f acc[4];
#pragma unroll
  for (int t = 0; t < 4; ++t) acc[t] = vzero();

  for (int kb = 0; kb < qb + 16; kb += 32) {
    // prefetch next key block (K rows + V rows) while this block computes
    int kn = kb + 32;
    if (kn < qb + 16) {
      __builtin_prefetch(Kp + (size_t)(kn + r) * Dk, 0, 3);
      __builtin_prefetch(Kp + (size_t)(kn + 16 + r) * Dk, 0, 3);
#pragma unroll
      for (int t = 0; t < 4; ++t)
        __builtin_prefetch(Vp + (size_t)(t * 16 + r) * Sq + kn, 0, 3);
    }

    v8f s0 = vzero(), s1 = vzero();
    {
      v16bf k00 = load_frag(Kp + (size_t)(kb + r) * Dk, half, 0);
      v16bf k01 = load_frag(Kp + (size_t)(kb + r) * Dk, half, 32);
      s0 = wmma_bf16(qa0, k00, s0);
      s0 = wmma_bf16(qa1, k01, s0);
      v16bf k10 = load_frag(Kp + (size_t)(kb + 16 + r) * Dk, half, 0);
      v16bf k11 = load_frag(Kp + (size_t)(kb + 16 + r) * Dk, half, 32);
      s1 = wmma_bf16(qa0, k10, s1);
      s1 = wmma_bf16(qa1, k11, s1);
    }
    int c0 = kb + r, c1 = kb + 16 + r;
    float p0[8], p1[8];
#pragma unroll
    for (int i = 0; i < 8; ++i) {
      int row = qb + 8 * half + i;
      // ref: scale by 1/sqrt(64) then add tril-mask value (+1 allowed, -inf masked)
      float v0 = s0[i] * 0.125f + 1.0f;
      float v1 = s1[i] * 0.125f + 1.0f;
      if (c0 > row) v0 = -__builtin_inff();
      if (c1 > row) v1 = -__builtin_inff();
      float m = fmaxf(v0, v1);
#pragma unroll
      for (int off = 1; off <= 8; off <<= 1)
        m = fmaxf(m, __shfl_xor(m, off, 32));      // reduce within 16-lane half
      float nm = fmaxf(rowmax[i], m);
      float fac = __expf(rowmax[i] - nm);
      float e0 = __expf(v0 - nm);
      float e1 = __expf(v1 - nm);
      float ss = e0 + e1;
#pragma unroll
      for (int off = 1; off <= 8; off <<= 1)
        ss += __shfl_xor(ss, off, 32);
      rowsum[i] = rowsum[i] * fac + ss;
      rowmax[i] = nm;
#pragma unroll
      for (int t = 0; t < 4; ++t) acc[t][i] *= fac;
      p0[i] = e0; p1[i] = e1;
    }
    // C-layout -> A-fragment layout via per-wave LDS tile (16x32 bf16)
    __bf16* lp = lds_p[wave];
#pragma unroll
    for (int i = 0; i < 8; ++i) {
      lp[(8 * half + i) * 32 + r]      = (__bf16)p0[i];
      lp[(8 * half + i) * 32 + 16 + r] = (__bf16)p1[i];
    }
    v16bf pa = load_frag(lp + r * 32, half, 0);
#pragma unroll
    for (int t = 0; t < 4; ++t) {
      v16bf vb = load_frag(Vp + (size_t)(t * 16 + r) * Sq, half, kb);
      acc[t] = wmma_bf16(pa, vb, acc[t]);
    }
  }

#pragma unroll
  for (int i = 0; i < 8; ++i) {
    int s = qb + 8 * half + i;
    float inv = 1.0f / rowsum[i];
    size_t base = ((size_t)(b * Sq + s)) * Dm + (size_t)h * Dk + r;
#pragma unroll
    for (int t = 0; t < 4; ++t)
      AO[base + t * 16] = (__bf16)(acc[t][i] * inv);
  }
}

// ---------------------------------------------------------------------------
// Output projection: out fp32 [8192,1024] = AObf16 @ WprojT' + bproj
// ---------------------------------------------------------------------------
__global__ __launch_bounds__(256) void gemm_proj(
    const __bf16* __restrict__ X, const __bf16* __restrict__ Wt,
    const float* __restrict__ bias, float* __restrict__ Out) {
  int lane = threadIdx.x & 31, wave = threadIdx.x >> 5;
  int r = lane & 15, half = lane >> 4;
  int m0 = blockIdx.y * 256 + wave * 32;
  int n0 = blockIdx.x * 64;

  v8f acc[2][4];
#pragma unroll
  for (int a = 0; a < 2; ++a)
#pragma unroll
    for (int t = 0; t < 4; ++t) acc[a][t] = vzero();

  gemm_core(X + (size_t)(m0 + r) * Dm, X + (size_t)(m0 + 16 + r) * Dm,
            Wt + (size_t)(n0 + r) * Dm, Wt + (size_t)(n0 + 16 + r) * Dm,
            Wt + (size_t)(n0 + 32 + r) * Dm, Wt + (size_t)(n0 + 48 + r) * Dm,
            half, acc);

#pragma unroll
  for (int mt = 0; mt < 2; ++mt) {
#pragma unroll
    for (int t = 0; t < 4; ++t) {
      int n = n0 + t * 16 + r;
      float bs = bias[n];
#pragma unroll
      for (int i = 0; i < 8; ++i) {
        int mm = m0 + mt * 16 + 8 * half + i;
        Out[(size_t)mm * Dm + n] = acc[mt][t][i] + bs;
      }
    }
  }
}

// ---------------------------------------------------------------------------
extern "C" void kernel_launch(void* const* d_in, const int* in_sizes, int n_in,
                              void* d_out, int out_size, void* d_ws,
                              size_t ws_size, hipStream_t stream) {
  const float* x     = (const float*)d_in[0];
  const float* Wqkv  = (const float*)d_in[1];
  const float* bqkv  = (const float*)d_in[2];
  const float* Wproj = (const float*)d_in[3];
  const float* bproj = (const float*)d_in[4];
  float* out = (float*)d_out;

  char* w = (char*)d_ws;
  const size_t MB = (size_t)1 << 20;
  __bf16* xb     = (__bf16*)(w + 0 * MB);    // 16 MB  x as bf16
  __bf16* wqkvt  = (__bf16*)(w + 16 * MB);   //  6 MB  Wqkv^T bf16 [3072,1024]
  __bf16* wprojt = (__bf16*)(w + 22 * MB);   //  2 MB  Wproj^T bf16 [1024,1024]
  __bf16* qws    = (__bf16*)(w + 24 * MB);   // 16 MB  Q [B,H,S,64]
  __bf16* kws    = (__bf16*)(w + 40 * MB);   // 16 MB  K [B,H,S,64]
  __bf16* vtws   = (__bf16*)(w + 56 * MB);   // 16 MB  V^T [B,H,64,S]
  __bf16* ao     = xb;                       // alias: xb dead after QKV GEMM

  const int nX = 2 * Sq * Dm;         // 8388608
  const int nWq = Dm * 3 * Dm;        // 3145728
  const int nWp = Dm * Dm;            // 1048576

  cvt_f32_bf16<<<(nX + 255) / 256, 256, 0, stream>>>(x, xb, nX);
  transpose_cvt<<<(nWq + 255) / 256, 256, 0, stream>>>(Wqkv, wqkvt, 3 * Dm, nWq);
  transpose_cvt<<<(nWp + 255) / 256, 256, 0, stream>>>(Wproj, wprojt, Dm, nWp);

  gemm_qkv<<<dim3(48, 32), 256, 0, stream>>>(xb, wqkvt, bqkv, qws, kws, vtws);
  attn_fwd<<<dim3(Sq / 128, 2 * Hh), 256, 0, stream>>>(qws, kws, vtws, ao);
  gemm_proj<<<dim3(16, 32), 256, 0, stream>>>(ao, wprojt, bproj, out);
}
